// ObjectDetection_47966194762036
// MI455X (gfx1250) — matrix operand order repaired and checked by reading, not compiled, not run on hardware
//
#include <hip/hip_runtime.h>
#include <hip/hip_bf16.h>

// ---------------------------------------------------------------------------
// CDNA5 (gfx1250, wave32) object-detection head.
// Conv3x3 implemented as implicit GEMM with V_WMMA_F32_16X16X32_F16.
// One wave computes a (MT*16 positions) x (NT*16 couts) tile: MT*NT v8f
// accumulators, A fragments reused NT times, B fragments reused MT times.
// Zero padding handled by redirecting OOB rows to a zeroed pad buffer
// (address cndmask instead of per-chunk EXEC masking).
// Activations: channels-last f16 [B, NPOS, 256]; weights: [tap][coutP][cin].
// ---------------------------------------------------------------------------

typedef __attribute__((ext_vector_type(16))) _Float16 v16h;
typedef __attribute__((ext_vector_type(8)))  _Float16 v8h;
typedef __attribute__((ext_vector_type(8)))  float    v8f;

#define NPOS   5456     // 64^2 + 32^2 + 16^2 + 8^2 + 4^2
#define NB     16       // batch
#define NCH    256
#define NCLS   80
#define EPSV   1e-5f
#define MAXI   100

// ---------------------------------------------------------------------------
// Fragment layouts (CDNA5 ISA 7.12.2):
// A (16x32 f16): lane L row M=L&15; vec elem j<8 -> K=8*(L>>4)+j,
//                j>=8 -> K=8*(L>>4)+16+(j-8).
// B (32x16 f16): lane L column N=L&15; K order as for A; this code loads both A and B as K=16*(L>>4)+j.
// C/D (16x16 f32): lane L -> N=L&15, VGPR v -> M=8*(L>>4)+v.
// ---------------------------------------------------------------------------
template <int MT, int NT>
__global__ __launch_bounds__(32) void k_conv3x3_wmma(
    const _Float16* __restrict__ act,   // [NB, NPOS, 256] f16 channels-last
    const _Float16* __restrict__ wgt,   // [9, coutP, 256] f16
    const float*    __restrict__ bias,  // [coutP] f32 (pre-padded)
    const _Float16* __restrict__ zpad,  // 256 zero halves
    float*          __restrict__ out,   // [NB, NPOS, coutP] f32
    int H, int base, int coutP, int relu)
{
  const int lane = threadIdx.x;
  const int mt = blockIdx.x, nt = blockIdx.y, bb = blockIdx.z;
  const int W = H;
  const int row = lane & 15;     // A row / B column within 16-tile
  const int kh  = lane >> 4;     // K-half selector

  v8f acc[MT][NT];
#pragma unroll
  for (int i = 0; i < MT; ++i)
#pragma unroll
    for (int u = 0; u < NT; ++u) acc[i][u] = (v8f)0.0f;

  int ys[MT], xs[MT];
#pragma unroll
  for (int i = 0; i < MT; ++i) {
    int p = (mt * MT + i) * 16 + row;
    ys[i] = p / W;
    xs[i] = p - ys[i] * W;
  }
  const size_t actBase = (size_t)bb * NPOS + (size_t)base;
  const _Float16* zp = zpad + kh * 16;

#pragma unroll 1
  for (int t = 0; t < 9; ++t) {
    const int dy = t / 3 - 1;
    const int dx = t - (t / 3) * 3 - 1;
    const _Float16* aps[MT];
#pragma unroll
    for (int i = 0; i < MT; ++i) {
      const int ny = ys[i] + dy, nx = xs[i] + dx;
      const bool inb = ((unsigned)ny < (unsigned)H) && ((unsigned)nx < (unsigned)W);
      const int q = ny * W + nx;
      aps[i] = inb ? (act + (actBase + (size_t)q) * NCH + kh * 16) : zp;
    }
    const _Float16* bps[NT];
#pragma unroll
    for (int u = 0; u < NT; ++u)
      bps[u] = wgt + ((size_t)t * coutP + (nt * NT + u) * 16 + row) * NCH + kh * 16;

#pragma unroll 1
    for (int ck = 0; ck < 8; ++ck) {          // 8 chunks of K=32 -> cin 0..255
      union { v16h v; v8h h[2]; } af[MT];
#pragma unroll
      for (int i = 0; i < MT; ++i) {
        af[i].h[0] = *(const v8h*)(aps[i] + ck * 32);        // K = 16*kh + 0..7
        af[i].h[1] = *(const v8h*)(aps[i] + ck * 32 + 8);    // K = 16*kh + 8..15
      }
      v16h bf[NT];
#pragma unroll
      for (int u = 0; u < NT; ++u)
        bf[u] = *(const v16h*)(bps[u] + ck * 32);            // K = 16*kh + 0..15
#pragma unroll
      for (int i = 0; i < MT; ++i)
#pragma unroll
        for (int u = 0; u < NT; ++u)
          acc[i][u] = __builtin_amdgcn_wmma_f32_16x16x32_f16(
              false, af[i].v, false, bf[u], (short)0, acc[i][u], false, false);
    }
  }

  const int n = lane & 15;
#pragma unroll
  for (int u = 0; u < NT; ++u) {
    const int cout = (nt * NT + u) * 16 + n;
    const float bs = bias[cout];
#pragma unroll
    for (int i = 0; i < MT; ++i) {
#pragma unroll
      for (int v = 0; v < 8; ++v) {
        const int M = kh * 8 + v;
        float val = acc[i][u][v] + bs;
        if (relu) val = fmaxf(val, 0.0f);
        out[((size_t)bb * NPOS + base + (mt * MT + i) * 16 + M) * (size_t)coutP + cout] = val;
      }
    }
  }
}

// NCHW f32 input -> channels-last f16 concatenated activation buffer
__global__ void k_cvt_input(const float* __restrict__ x, _Float16* __restrict__ act,
                            int H, int base)
{
  const int HW = H * H;
  const int total = NB * NCH * HW;
  int idx = blockIdx.x * blockDim.x + threadIdx.x;
  if (idx >= total) return;
  int b = idx / (NCH * HW);
  int r = idx - b * NCH * HW;
  int c = r / HW;
  int p = r - c * HW;
  act[((size_t)b * NPOS + base + p) * NCH + c] = (_Float16)x[idx];
}

// w f32 [O,256,3,3] -> f16 [9][coutP][256], zero-padded for cout>=O
__global__ void k_prep_w(const float* __restrict__ w, _Float16* __restrict__ out,
                         int O, int coutP)
{
  const int total = 9 * coutP * NCH;
  int idx = blockIdx.x * blockDim.x + threadIdx.x;
  if (idx >= total) return;
  int t = idx / (coutP * NCH);
  int r = idx - t * coutP * NCH;
  int o = r / NCH;
  int i = r - o * NCH;
  out[idx] = (o < O) ? (_Float16)w[((size_t)o * NCH + i) * 9 + t] : (_Float16)0.0f;
}

__global__ void k_pad_bias(const float* __restrict__ b, float* __restrict__ out,
                           int O, int coutP)
{
  int i = blockIdx.x * blockDim.x + threadIdx.x;
  if (i < coutP) out[i] = (i < O) ? b[i] : 0.0f;
}

__global__ void k_zero_f16(_Float16* __restrict__ p, int n)
{
  int i = blockIdx.x * blockDim.x + threadIdx.x;
  if (i < n) p[i] = (_Float16)0.0f;
}

// GroupNorm stats per (group, batch) for one level: mean/var over (8 ch x HW)
__global__ __launch_bounds__(256) void k_gn_stats(
    const float* __restrict__ xin, float* __restrict__ stats, int HW, int base)
{
  __shared__ float ss[256], sq[256];
  const int g = blockIdx.x, b = blockIdx.y, tid = threadIdx.x;
  const int n = HW * 8;
  float s = 0.0f, q = 0.0f;
  const float* bp = xin + ((size_t)b * NPOS + base) * NCH + g * 8;
  for (int i = tid; i < n; i += 256) {
    int p = i >> 3, j = i & 7;
    float v = bp[(size_t)p * NCH + j];
    s += v; q += v * v;
  }
  ss[tid] = s; sq[tid] = q;
  __syncthreads();
  for (int st = 128; st > 0; st >>= 1) {
    if (tid < st) { ss[tid] += ss[tid + st]; sq[tid] += sq[tid + st]; }
    __syncthreads();
  }
  if (tid == 0) {
    float m = ss[0] / (float)n;
    float var = sq[0] / (float)n - m * m;
    stats[(b * 32 + g) * 2 + 0] = m;
    stats[(b * 32 + g) * 2 + 1] = var;
  }
}

// normalize + affine + ReLU -> f16 channels-last
__global__ void k_gn_norm(const float* __restrict__ xin, const float* __restrict__ stats,
                          const float* __restrict__ gamma, const float* __restrict__ beta,
                          _Float16* __restrict__ dst, int HW, int base)
{
  const int total = NB * HW * NCH;
  int idx = blockIdx.x * blockDim.x + threadIdx.x;
  if (idx >= total) return;
  int b = idx / (HW * NCH);
  int r = idx - b * HW * NCH;
  int p = r / NCH;
  int c = r - p * NCH;
  int g = c >> 3;
  float m = stats[(b * 32 + g) * 2 + 0];
  float var = stats[(b * 32 + g) * 2 + 1];
  size_t o = ((size_t)b * NPOS + base + p) * NCH + c;
  float v = (xin[o] - m) * rsqrtf(var + EPSV) * gamma[c] + beta[c];
  dst[o] = (_Float16)fmaxf(v, 0.0f);
}

// sigmoid-max/argmax over 80 classes + box decode
__global__ void k_decode(const float* __restrict__ logits, const float* __restrict__ boxraw,
                         float* __restrict__ scores, int* __restrict__ classes,
                         float* __restrict__ boxes)
{
  int idx = blockIdx.x * blockDim.x + threadIdx.x;
  if (idx >= NB * NPOS) return;
  const int p = idx - (idx / NPOS) * NPOS;
  const float* lp = logits + (size_t)idx * NCLS;
  float best = -1e30f; int bi = 0;
  for (int j = 0; j < NCLS; ++j) {
    float z = lp[j];
    if (z > best) { best = z; bi = j; }
  }
  scores[idx] = 1.0f / (1.0f + expf(-best));
  classes[idx] = bi;

  int lvl, base, H;
  if      (p < 4096) { lvl = 3; base = 0;    H = 64; }
  else if (p < 5120) { lvl = 4; base = 4096; H = 32; }
  else if (p < 5376) { lvl = 5; base = 5120; H = 16; }
  else if (p < 5440) { lvl = 6; base = 5376; H = 8;  }
  else               { lvl = 7; base = 5440; H = 4;  }
  const float stride = (float)(1 << lvl);
  int lp2 = p - base;
  int y = lp2 / H, x = lp2 - y * H;
  float gx = (x + 0.5f) * stride, gy = (y + 0.5f) * stride;
  const float* br = boxraw + (size_t)idx * 16;   // padded cout stride 16
  float* ob = boxes + (size_t)idx * 4;
  ob[0] = gx - (stride * br[0] + 1.0f);
  ob[1] = gy - (stride * br[1] + 1.0f);
  ob[2] = gx + (stride * br[2] + 1.0f);
  ob[3] = gy + (stride * br[3] + 1.0f);
}

// Per-image class-agnostic hard NMS: iterative argmax-select (score desc,
// index asc on ties) + IoU>0.6 suppression, top-100 kept, pad zeros.
__global__ __launch_bounds__(256) void k_nms(
    const float* __restrict__ scores, const int* __restrict__ classes,
    const float* __restrict__ boxes, float* __restrict__ out)
{
  __shared__ float rs[256];
  __shared__ int ri[256];
  __shared__ unsigned char sup[NPOS];
  __shared__ float bx[4];
  __shared__ int done;

  const int b = blockIdx.x, tid = threadIdx.x;
  const float* sc = scores + (size_t)b * NPOS;
  const int*   cl = classes + (size_t)b * NPOS;
  const float* bxs = boxes + (size_t)b * NPOS * 4;

  float* out_num = out;
  float* out_s = out + NB;
  float* out_c = out + NB + NB * MAXI;
  float* out_b = out + NB + 2 * NB * MAXI;

  for (int i = tid; i < MAXI; i += 256) { out_s[b * MAXI + i] = 0.0f; out_c[b * MAXI + i] = 0.0f; }
  for (int i = tid; i < 4 * MAXI; i += 256) out_b[b * 4 * MAXI + i] = 0.0f;
  for (int p = tid; p < NPOS; p += 256) sup[p] = (sc[p] > 0.05f) ? 0 : 1;
  if (tid == 0) done = 0;
  __syncthreads();

  for (int k = 0; k < MAXI; ++k) {
    float bs = -1e30f; int bi = -1;
    for (int p = tid; p < NPOS; p += 256) {
      if (!sup[p]) {
        float s = sc[p];
        if (s > bs || (s == bs && (bi < 0 || p < bi))) { bs = s; bi = p; }
      }
    }
    rs[tid] = bs; ri[tid] = bi;
    __syncthreads();
    for (int st = 128; st > 0; st >>= 1) {
      if (tid < st) {
        float so = rs[tid + st]; int io = ri[tid + st];
        if (io >= 0 && (ri[tid] < 0 || so > rs[tid] || (so == rs[tid] && io < ri[tid]))) {
          rs[tid] = so; ri[tid] = io;
        }
      }
      __syncthreads();
    }
    if (tid == 0) {
      int best = ri[0];
      if (best < 0) { done = 1; }
      else {
        out_s[b * MAXI + k] = rs[0];
        out_c[b * MAXI + k] = (float)cl[best];
        bx[0] = bxs[best * 4 + 0]; bx[1] = bxs[best * 4 + 1];
        bx[2] = bxs[best * 4 + 2]; bx[3] = bxs[best * 4 + 3];
        out_b[(b * MAXI + k) * 4 + 0] = bx[0];
        out_b[(b * MAXI + k) * 4 + 1] = bx[1];
        out_b[(b * MAXI + k) * 4 + 2] = bx[2];
        out_b[(b * MAXI + k) * 4 + 3] = bx[3];
        sup[best] = 1;
      }
    }
    __syncthreads();
    if (done) break;
    const float ax1 = bx[0], ay1 = bx[1], ax2 = bx[2], ay2 = bx[3];
    const float aarea = (ax2 - ax1) * (ay2 - ay1);
    for (int p = tid; p < NPOS; p += 256) {
      if (!sup[p]) {
        float x1 = fmaxf(ax1, bxs[p * 4 + 0]);
        float y1 = fmaxf(ay1, bxs[p * 4 + 1]);
        float x2 = fminf(ax2, bxs[p * 4 + 2]);
        float y2 = fminf(ay2, bxs[p * 4 + 3]);
        float inter = fmaxf(x2 - x1, 0.0f) * fmaxf(y2 - y1, 0.0f);
        float barea = (bxs[p * 4 + 2] - bxs[p * 4 + 0]) * (bxs[p * 4 + 3] - bxs[p * 4 + 1]);
        float iou = inter / (aarea + barea - inter);
        if (iou > 0.6f) sup[p] = 1;
      }
    }
    __syncthreads();
  }
  __syncthreads();
  if (tid == 0) {
    int cnt = 0;
    for (int i = 0; i < MAXI; ++i) if (out_s[b * MAXI + i] > 0.5f) cnt++;
    out_num[b] = (float)cnt;
  }
}

// ---------------------------------------------------------------------------
extern "C" void kernel_launch(void* const* d_in, const int* in_sizes, int n_in,
                              void* d_out, int out_size, void* d_ws, size_t ws_size,
                              hipStream_t stream)
{
  (void)in_sizes; (void)n_in; (void)out_size; (void)ws_size;
  static const int Hs[5]    = {64, 32, 16, 8, 4};
  static const int bases[5] = {0, 4096, 5120, 5376, 5440};

  // -------- inputs (JAX dict tree-flatten order: sorted keys) --------
  const float* xin[5];
  for (int l = 0; l < 5; ++l) xin[l] = (const float*)d_in[l];
  // cls_params: 'b'(final), blocks[i]{'b','bt','g','w'}, 'w'(final)
  const float* cls_bfin = (const float*)d_in[5];
  const float *cls_bB[4], *cls_bBt[4], *cls_bG[4], *cls_bW[4];
  for (int i = 0; i < 4; ++i) {
    cls_bB[i]  = (const float*)d_in[6 + 4 * i];
    cls_bBt[i] = (const float*)d_in[7 + 4 * i];
    cls_bG[i]  = (const float*)d_in[8 + 4 * i];
    cls_bW[i]  = (const float*)d_in[9 + 4 * i];
  }
  const float* cls_wfin = (const float*)d_in[22];
  const float* box_bfin = (const float*)d_in[23];
  const float *box_bB[4], *box_bBt[4], *box_bG[4], *box_bW[4];
  for (int i = 0; i < 4; ++i) {
    box_bB[i]  = (const float*)d_in[24 + 4 * i];
    box_bBt[i] = (const float*)d_in[25 + 4 * i];
    box_bG[i]  = (const float*)d_in[26 + 4 * i];
    box_bW[i]  = (const float*)d_in[27 + 4 * i];
  }
  const float* box_wfin = (const float*)d_in[40];

  // -------- workspace bump allocator (256B aligned) --------
  size_t off = 0;
  auto walloc = [&](size_t bytes) -> char* {
    char* p = (char*)d_ws + off;
    off += (bytes + 255) & ~(size_t)255;
    return p;
  };
  const size_t actElems = (size_t)NB * NPOS * NCH;
  _Float16* wcls   = (_Float16*)walloc(4 * 9 * (size_t)NCH * NCH * 2);
  _Float16* wbox   = (_Float16*)walloc(4 * 9 * (size_t)NCH * NCH * 2);
  _Float16* wclsf  = (_Float16*)walloc(9 * (size_t)NCLS * NCH * 2);
  _Float16* wboxf  = (_Float16*)walloc(9 * (size_t)16 * NCH * 2);
  float*    bclsf  = (float*)walloc(NCLS * 4);
  float*    bboxf  = (float*)walloc(16 * 4);
  _Float16* zpad   = (_Float16*)walloc(NCH * 2);
  _Float16* actA   = (_Float16*)walloc(actElems * 2);
  _Float16* actB   = (_Float16*)walloc(actElems * 2);
  float*    conv32 = (float*)walloc(actElems * 4);
  float*    stats  = (float*)walloc((size_t)NB * 32 * 2 * 4);
  float*    logits = (float*)walloc((size_t)NB * NPOS * NCLS * 4);
  float*    boxraw = (float*)walloc((size_t)NB * NPOS * 16 * 4);
  float*    scoreb = (float*)walloc((size_t)NB * NPOS * 4);
  int*      classb = (int*)walloc((size_t)NB * NPOS * 4);
  float*    boxesb = (float*)walloc((size_t)NB * NPOS * 4 * 4);

  // -------- weight / bias / zero-pad prep --------
  for (int i = 0; i < 4; ++i) {
    int tot = 9 * NCH * NCH;
    k_prep_w<<<(tot + 255) / 256, 256, 0, stream>>>(cls_bW[i], wcls + (size_t)i * tot, NCH, NCH);
    k_prep_w<<<(tot + 255) / 256, 256, 0, stream>>>(box_bW[i], wbox + (size_t)i * tot, NCH, NCH);
  }
  {
    int tot = 9 * NCLS * NCH;
    k_prep_w<<<(tot + 255) / 256, 256, 0, stream>>>(cls_wfin, wclsf, NCLS, NCLS);
    tot = 9 * 16 * NCH;
    k_prep_w<<<(tot + 255) / 256, 256, 0, stream>>>(box_wfin, wboxf, 4, 16);
    k_pad_bias<<<1, 256, 0, stream>>>(cls_bfin, bclsf, NCLS, NCLS);
    k_pad_bias<<<1, 256, 0, stream>>>(box_bfin, bboxf, 4, 16);
    k_zero_f16<<<1, 256, 0, stream>>>(zpad, NCH);
  }

  // -------- conv dispatch: pick tile shape per level / head --------
  auto launch_conv = [&](const _Float16* src, const _Float16* w, const float* bias,
                         float* outp, int l, int coutP, int relu) {
    const int H = Hs[l], HW = H * H, base = bases[l];
    const bool bigM = (HW % 64) == 0;   // levels 3..6
    if (coutP == NCH) {
      if (bigM) {
        dim3 g(HW / 64, NCH / 64, NB);
        k_conv3x3_wmma<4, 4><<<g, 32, 0, stream>>>(src, w, bias, zpad, outp, H, base, coutP, relu);
      } else {
        dim3 g(HW / 16, NCH / 64, NB);
        k_conv3x3_wmma<1, 4><<<g, 32, 0, stream>>>(src, w, bias, zpad, outp, H, base, coutP, relu);
      }
    } else {
      if (bigM) {
        dim3 g(HW / 64, coutP / 16, NB);
        k_conv3x3_wmma<4, 1><<<g, 32, 0, stream>>>(src, w, bias, zpad, outp, H, base, coutP, relu);
      } else {
        dim3 g(HW / 16, coutP / 16, NB);
        k_conv3x3_wmma<1, 1><<<g, 32, 0, stream>>>(src, w, bias, zpad, outp, H, base, coutP, relu);
      }
    }
  };

  // -------- run one head (4 conv-GN-relu blocks + final conv) --------
  auto run_head = [&](const _Float16* wblk,
                      const float* const* bB, const float* const* bG, const float* const* bBt,
                      const _Float16* wfin, const float* bfinPad, int coutPfin,
                      float* outbuf, int final_relu) {
    for (int l = 0; l < 5; ++l) {
      int H = Hs[l], HW = H * H, tot = NB * NCH * HW;
      k_cvt_input<<<(tot + 255) / 256, 256, 0, stream>>>(xin[l], actA, H, bases[l]);
    }
    _Float16* src = actA;
    _Float16* dst = actB;
    for (int L = 0; L < 4; ++L) {
      const _Float16* wl = wblk + (size_t)L * 9 * NCH * NCH;
      for (int l = 0; l < 5; ++l) {
        int H = Hs[l], HW = H * H;
        launch_conv(src, wl, bB[L], conv32, l, NCH, 0);
        dim3 gs(32, NB);
        k_gn_stats<<<gs, 256, 0, stream>>>(conv32, stats, HW, bases[l]);
        int tot = NB * HW * NCH;
        k_gn_norm<<<(tot + 255) / 256, 256, 0, stream>>>(conv32, stats, bG[L], bBt[L],
                                                         dst, HW, bases[l]);
      }
      _Float16* t = src; src = dst; dst = t;
    }
    for (int l = 0; l < 5; ++l)
      launch_conv(src, wfin, bfinPad, outbuf, l, coutPfin, final_relu);
  };

  run_head(wcls, cls_bB, cls_bG, cls_bBt, wclsf, bclsf, NCLS, logits, 0);
  run_head(wbox, box_bB, box_bG, box_bBt, wboxf, bboxf, 16, boxraw, 1);

  // -------- decode + NMS --------
  {
    int tot = NB * NPOS;
    k_decode<<<(tot + 255) / 256, 256, 0, stream>>>(logits, boxraw, scoreb, classb, boxesb);
  }
  k_nms<<<NB, 256, 0, stream>>>(scoreb, classb, boxesb, (float*)d_out);
}
